// LSTM_11871289606766
// MI455X (gfx1250) — matrix operand order repaired and checked by reading, not compiled, not run on hardware
//
#include <hip/hip_runtime.h>
#include <hip/hip_bf16.h>

// ---------------------------------------------------------------------------
// LSTM (B=128, T=512, D=512, H=1024, O=256) for gfx1250 (MI455X).
//   * W_lstm / W_out packed once into bf16 WMMA B-fragment layout (32B/lane).
//   * One fused kernel per timestep: each wave owns a 16x16 tile of H columns
//     and computes the i/j/f/o gate quartet (4 WMMA accumulators, shared A
//     fragment), double-buffered fragment loads so WMMAs overlap loads,
//     then fused sigmoid/tanh gating + seq_len masking + bf16 h ping-pong.
//   * Projection = double-buffered bf16 WMMA GEMM [65536,1024]x[1024,256].
// Weights (12.6 MB bf16) stay resident in the 192 MB L2 across 512 steps.
// ---------------------------------------------------------------------------

#define B_  128
#define T_  512
#define D_  512
#define H_  1024
#define O_  256
#define G4H 4096
#define NKT  48               // (D+H)/32 k-tiles for the LSTM gemm
#define NKT_X 16              // k-tiles covering the x part (D/32)
#define NKT_P 32              // k-tiles for the projection (H/32)

typedef __attribute__((ext_vector_type(16))) __bf16 bf16x16;
typedef __attribute__((ext_vector_type(8)))  float  f32x8;

union FragB16 { bf16x16 v; uint4 u[2]; };

__device__ __forceinline__ float sigmoidf_(float x) {
    return 1.0f / (1.0f + __expf(-x));
}
__device__ __forceinline__ float tanhf_(float x) {
    // tanh(x) = sign(x) * (1 - e) / (1 + e), e = exp(-2|x|)  (stable for all x)
    float e = __expf(-2.0f * fabsf(x));
    float r = (1.0f - e) / (1.0f + e);
    return copysignf(r, x);
}

// --- workspace layout (bytes) ----------------------------------------------
#define WS_WP    0u                                  // 48*256*512 bf16 = 12,582,912 B
#define WS_WOUT  12582912u                           // 32*16*512 bf16  =    524,288 B
#define WS_CST   12845056u                           // c state fp32    =    524,288 B
#define WS_HST   13369344u                           // h state fp32    =    524,288 B
#define WS_HBF0  13893632u                           // h bf16 buf 0    =    262,144 B
#define WS_HBF1  14155776u                           // h bf16 buf 1    =    262,144 B

// --- d_out layout (fp32 elements) ------------------------------------------
#define OUT_LOGITS 0u
#define OUT_LSTM   16777216u
#define OUT_COUT   83886080u
#define OUT_HOUT   84017152u

// ---------------------------------------------------------------------------
__global__ void pack_w_lstm(const float* __restrict__ W, __bf16* __restrict__ Wp) {
    unsigned idx = blockIdx.x * 256u + threadIdx.x;      // 6,291,456 elements
    unsigned i    = idx & 15u;
    unsigned lane = (idx >> 4) & 31u;
    unsigned nt   = (idx >> 9) & 255u;
    unsigned kt   = idx >> 17;
    unsigned krow = kt * 32u + ((lane >= 16u) ? 8u : 0u) + (i & 7u) + ((i >= 8u) ? 16u : 0u);
    unsigned col  = nt * 16u + (lane & 15u);
    Wp[idx] = (__bf16)W[krow * (unsigned)G4H + col];
}

__global__ void pack_w_out(const float* __restrict__ W, __bf16* __restrict__ Wp) {
    unsigned idx = blockIdx.x * 256u + threadIdx.x;      // 262,144 elements
    unsigned i    = idx & 15u;
    unsigned lane = (idx >> 4) & 31u;
    unsigned nt   = (idx >> 9) & 15u;
    unsigned kt   = idx >> 13;
    unsigned krow = kt * 32u + ((lane >= 16u) ? 8u : 0u) + (i & 7u) + ((i >= 8u) ? 16u : 0u);
    unsigned col  = nt * 16u + (lane & 15u);
    Wp[idx] = (__bf16)W[krow * (unsigned)O_ + col];
}

__global__ void init_state(const float* __restrict__ c_in, const float* __restrict__ h_in,
                           float* __restrict__ c_st, float* __restrict__ h_st,
                           __bf16* __restrict__ hbf0) {
    unsigned idx = blockIdx.x * 256u + threadIdx.x;      // 131,072
    float c = c_in[idx], h = h_in[idx];
    c_st[idx] = c;
    h_st[idx] = h;
    hbf0[idx] = (__bf16)h;
}

// --- fragment load helpers --------------------------------------------------
__device__ __forceinline__ void loadA_f32(FragB16& a, const float* p) {
    #pragma unroll
    for (int i = 0; i < 8; ++i) a.v[i]     = (__bf16)p[i];
    #pragma unroll
    for (int i = 0; i < 8; ++i) a.v[8 + i] = (__bf16)p[16 + i];
}
__device__ __forceinline__ void loadA_bf16(FragB16& a, const __bf16* p) {
    a.u[0] = *(const uint4*)(p);
    a.u[1] = *(const uint4*)(p + 16);
}
__device__ __forceinline__ void loadB1(FragB16& b, const __bf16* p) {
    b.u[0] = *(const uint4*)(p);
    b.u[1] = *(const uint4*)(p + 8);
}

#define WMMA_BF16(ACC, A, Bv) \
    ACC = __builtin_amdgcn_wmma_f32_16x16x32_bf16(false, (A).v, false, (Bv).v, (short)0, ACC, false, false)

// ---------------------------------------------------------------------------
// One LSTM timestep. 512 waves: q -> mt = q>>6 (8 batch row tiles),
// nh = q&63 (64 H column tiles). Double-buffered fragments: loads for k-tile
// kt+1 are issued before the 4 WMMAs of kt, so math overlaps memory.
// ---------------------------------------------------------------------------
__global__ void __launch_bounds__(128)
lstm_step(const float* __restrict__ inputs, const float* __restrict__ b_lstm,
          const int* __restrict__ seq_lens, const __bf16* __restrict__ Wp,
          const __bf16* __restrict__ h_prev_bf, __bf16* __restrict__ h_next_bf,
          float* __restrict__ c_st, float* __restrict__ h_st,
          float* __restrict__ lstm_out, int t) {
    const unsigned lane = threadIdx.x & 31u;
    const unsigned wave = threadIdx.x >> 5;
    const unsigned q    = blockIdx.x * 4u + wave;
    const unsigned mt   = q >> 6;
    const unsigned nh   = q & 63u;

    const unsigned arow = mt * 16u + (lane & 15u);
    const unsigned ksub = (lane >= 16u) ? 8u : 0u;

    const float*  xrow = inputs + ((size_t)arow * T_ + (unsigned)t) * D_ + ksub;
    const __bf16* hrow = h_prev_bf + (size_t)arow * H_ + ksub;
    const __bf16* wbase = Wp + (size_t)nh * 512u + lane * 16u;   // block (0, nh)
    const size_t KT_STRIDE   = 256u * 512u;   // elements between k-tiles
    const size_t GATE_STRIDE = 64u * 512u;    // elements between gates

    f32x8 acc0 = {}, acc1 = {}, acc2 = {}, acc3 = {};
    FragB16 a, b0, b1, b2, b3;

    // prologue: k-tile 0 (x part)
    loadA_f32(a, xrow);
    {
        const __bf16* bp = wbase;
        loadB1(b0, bp);
        loadB1(b1, bp + GATE_STRIDE);
        loadB1(b2, bp + 2 * GATE_STRIDE);
        loadB1(b3, bp + 3 * GATE_STRIDE);
    }

    // ---- x portion: kt = 0..14, next tile also from x --------------------
    #pragma unroll 5
    for (unsigned kt = 0; kt < 15; ++kt) {
        FragB16 an, c0, c1, c2, c3;
        loadA_f32(an, xrow + (kt + 1) * 32u);
        const __bf16* bp = wbase + (size_t)(kt + 1) * KT_STRIDE;
        __builtin_prefetch(bp + 2 * KT_STRIDE, 0, 1);
        loadB1(c0, bp);
        loadB1(c1, bp + GATE_STRIDE);
        loadB1(c2, bp + 2 * GATE_STRIDE);
        loadB1(c3, bp + 3 * GATE_STRIDE);
        WMMA_BF16(acc0, a, b0);
        WMMA_BF16(acc1, a, b1);
        WMMA_BF16(acc2, a, b2);
        WMMA_BF16(acc3, a, b3);
        a = an; b0 = c0; b1 = c1; b2 = c2; b3 = c3;
    }

    // ---- bridge: kt = 15 computes, next tile (16) comes from h -----------
    {
        FragB16 an, c0, c1, c2, c3;
        loadA_bf16(an, hrow);
        const __bf16* bp = wbase + (size_t)16 * KT_STRIDE;
        loadB1(c0, bp);
        loadB1(c1, bp + GATE_STRIDE);
        loadB1(c2, bp + 2 * GATE_STRIDE);
        loadB1(c3, bp + 3 * GATE_STRIDE);
        WMMA_BF16(acc0, a, b0);
        WMMA_BF16(acc1, a, b1);
        WMMA_BF16(acc2, a, b2);
        WMMA_BF16(acc3, a, b3);
        a = an; b0 = c0; b1 = c1; b2 = c2; b3 = c3;
    }

    // ---- h portion: kt = 16..46, next tile from h ------------------------
    #pragma unroll 4
    for (unsigned kt = 16; kt < 47; ++kt) {
        FragB16 an, c0, c1, c2, c3;
        loadA_bf16(an, hrow + (kt - 15) * 32u);
        const __bf16* bp = wbase + (size_t)(kt + 1) * KT_STRIDE;
        __builtin_prefetch(bp + 2 * KT_STRIDE, 0, 1);
        loadB1(c0, bp);
        loadB1(c1, bp + GATE_STRIDE);
        loadB1(c2, bp + 2 * GATE_STRIDE);
        loadB1(c3, bp + 3 * GATE_STRIDE);
        WMMA_BF16(acc0, a, b0);
        WMMA_BF16(acc1, a, b1);
        WMMA_BF16(acc2, a, b2);
        WMMA_BF16(acc3, a, b3);
        a = an; b0 = c0; b1 = c1; b2 = c2; b3 = c3;
    }

    // ---- tail: kt = 47 ----------------------------------------------------
    WMMA_BF16(acc0, a, b0);
    WMMA_BF16(acc1, a, b1);
    WMMA_BF16(acc2, a, b2);
    WMMA_BF16(acc3, a, b3);

    // ---- fused gating epilogue -------------------------------------------
    const unsigned col = nh * 16u + (lane & 15u);
    const float bi = b_lstm[col];
    const float bj = b_lstm[H_ + col];
    const float bf = b_lstm[2 * H_ + col];
    const float bo = b_lstm[3 * H_ + col];
    const unsigned rbase = mt * 16u + ((lane >> 4) << 3);

    #pragma unroll
    for (int r = 0; r < 8; ++r) {
        const unsigned row = rbase + r;
        const size_t sidx  = (size_t)row * H_ + col;
        const float ig = acc0[r] + bi;
        const float jg = acc1[r] + bj;
        const float fg = acc2[r] + bf;
        const float og = acc3[r] + bo;
        const float c_prev = c_st[sidx];
        const float h_prev = h_st[sidx];
        const float new_c  = c_prev * sigmoidf_(fg + 1.0f) + sigmoidf_(ig) * tanhf_(jg);
        const float new_h  = tanhf_(new_c) * sigmoidf_(og);
        const bool valid   = t < seq_lens[row];
        const float c2 = valid ? new_c : c_prev;
        const float h2 = valid ? new_h : h_prev;
        c_st[sidx] = c2;
        h_st[sidx] = h2;
        h_next_bf[sidx] = (__bf16)h2;
        lstm_out[((size_t)row * T_ + (unsigned)t) * H_ + col] = valid ? new_h : 0.0f;
    }
}

// ---------------------------------------------------------------------------
// Projection: logits = lstm_out @ W_out + b_out. 65536 waves, one 16x16 tile
// each (mt = q>>4 of 4096 row tiles, nt = q&15), K = 1024, double-buffered.
// ---------------------------------------------------------------------------
__global__ void __launch_bounds__(256)
proj_gemm(const float* __restrict__ lstm_out, const __bf16* __restrict__ WoutP,
          const float* __restrict__ b_out, float* __restrict__ logits) {
    const unsigned lane = threadIdx.x & 31u;
    const unsigned wave = threadIdx.x >> 5;
    const unsigned q    = blockIdx.x * 8u + wave;
    const unsigned mt   = q >> 4;
    const unsigned nt   = q & 15u;

    const unsigned arow = mt * 16u + (lane & 15u);
    const unsigned ksub = (lane >= 16u) ? 8u : 0u;
    const float*  xrow  = lstm_out + (size_t)arow * H_ + ksub;
    const __bf16* wbase = WoutP + (size_t)nt * 512u + lane * 16u;
    const size_t KT_STRIDE = 16u * 512u;

    f32x8 acc = {};
    FragB16 a, b;
    loadA_f32(a, xrow);
    loadB1(b, wbase);

    #pragma unroll 4
    for (unsigned kt = 0; kt < 31; ++kt) {
        FragB16 an, bn;
        loadA_f32(an, xrow + (kt + 1) * 32u);
        loadB1(bn, wbase + (size_t)(kt + 1) * KT_STRIDE);
        WMMA_BF16(acc, a, b);
        a = an; b = bn;
    }
    WMMA_BF16(acc, a, b);

    const unsigned col = nt * 16u + (lane & 15u);
    const float bb = b_out[col];
    const unsigned rbase = mt * 16u + ((lane >> 4) << 3);
    #pragma unroll
    for (int r = 0; r < 8; ++r) {
        logits[(size_t)(rbase + r) * O_ + col] = acc[r] + bb;
    }
}

__global__ void finalize_state(const float* __restrict__ c_st, const float* __restrict__ h_st,
                               float* __restrict__ c_out, float* __restrict__ h_out) {
    unsigned idx = blockIdx.x * 256u + threadIdx.x;      // 131,072
    c_out[idx] = c_st[idx];
    h_out[idx] = h_st[idx];
}

// ---------------------------------------------------------------------------
extern "C" void kernel_launch(void* const* d_in, const int* in_sizes, int n_in,
                              void* d_out, int out_size, void* d_ws, size_t ws_size,
                              hipStream_t stream) {
    const float* inputs   = (const float*)d_in[0];
    const float* c_in     = (const float*)d_in[1];
    const float* h_in     = (const float*)d_in[2];
    const float* W_lstm   = (const float*)d_in[3];
    const float* b_lstm   = (const float*)d_in[4];
    const float* W_out    = (const float*)d_in[5];
    const float* b_out    = (const float*)d_in[6];
    const int*   seq_lens = (const int*)d_in[7];

    char* ws = (char*)d_ws;
    __bf16* Wp    = (__bf16*)(ws + WS_WP);
    __bf16* WoutP = (__bf16*)(ws + WS_WOUT);
    float*  c_st  = (float*)(ws + WS_CST);
    float*  h_st  = (float*)(ws + WS_HST);
    __bf16* hbf[2] = { (__bf16*)(ws + WS_HBF0), (__bf16*)(ws + WS_HBF1) };

    float* out      = (float*)d_out;
    float* logits   = out + OUT_LOGITS;
    float* lstm_out = out + OUT_LSTM;
    float* c_out    = out + OUT_COUT;
    float* h_out    = out + OUT_HOUT;

    pack_w_lstm<<<24576, 256, 0, stream>>>(W_lstm, Wp);
    pack_w_out<<<1024, 256, 0, stream>>>(W_out, WoutP);
    init_state<<<512, 256, 0, stream>>>(c_in, h_in, c_st, h_st, hbf[0]);

    for (int t = 0; t < T_; ++t) {
        lstm_step<<<128, 128, 0, stream>>>(inputs, b_lstm, seq_lens, Wp,
                                           hbf[t & 1], hbf[(t + 1) & 1],
                                           c_st, h_st, lstm_out, t);
    }

    proj_gemm<<<8192, 256, 0, stream>>>(lstm_out, WoutP, b_out, logits);
    finalize_state<<<512, 256, 0, stream>>>(c_st, h_st, c_out, h_out);
}
